// EMTransformerLayer_23622320128493
// MI455X (gfx1250) — matrix operand order repaired and checked by reading, not compiled, not run on hardware
//
#include <hip/hip_runtime.h>
#include <math.h>

#define BB  2
#define TT  2048
#define DM  768
#define NH  12
#define DH  64
#define MM  (BB * TT)     // 4096 rows
#define DFF (4 * DM)      // 3072

typedef _Float16 h16v __attribute__((ext_vector_type(16)));
typedef _Float16 h8v  __attribute__((ext_vector_type(8)));
typedef float    f8v  __attribute__((ext_vector_type(8)));

__device__ __forceinline__ f8v zero8() {
  f8v v;
#pragma unroll
  for (int i = 0; i < 8; ++i) v[i] = 0.0f;
  return v;
}

__device__ __forceinline__ h16v combine16(h8v lo, h8v hi) {
  h16v a;
#pragma unroll
  for (int i = 0; i < 8; ++i) { a[i] = lo[i]; a[i + 8] = hi[i]; }
  return a;
}

// A-fragment 16x32 f16 (row-major source, stride in halves).
// lane 0..15: row=lane,    K = {0..15}
// lane 16..31: row=lane-16, K = {16..31}
__device__ __forceinline__ h16v load_a_frag(const _Float16* __restrict__ base,
                                            size_t stride, int row, int k0, int lane) {
  const int kb = (lane >> 4) * 16;
  const _Float16* p = base + (size_t)row * stride + (size_t)(k0 + kb);
  h8v lo = *(const h8v*)p;
  h8v hi = *(const h8v*)(p + 8);
  return combine16(lo, hi);
}

// B-fragment 32x16 f16 from K-major storage src[n][k] (B[k][n] = src[n][k]).
// lane: col = lane&15, K = 16*(lane>=16) + 0..15 -> one contiguous 32B load.
__device__ __forceinline__ h16v load_bt_frag(const _Float16* __restrict__ base,
                                             size_t stride, int col, int k0, int lane) {
  const int kb = (lane >> 4) * 16;
  return *(const h16v*)(base + (size_t)col * stride + (size_t)(k0 + kb));
}

__device__ __forceinline__ f8v wmma_f16(h16v a, h16v b, f8v c) {
  return __builtin_amdgcn_wmma_f32_16x16x32_f16(false, a, false, b, (short)0, c,
                                                false, false);
}

__device__ __forceinline__ float redmax16(float v) {
#pragma unroll
  for (int m = 1; m < 16; m <<= 1) v = fmaxf(v, __shfl_xor(v, m, 32));
  return v;
}
__device__ __forceinline__ float redsum16(float v) {
#pragma unroll
  for (int m = 1; m < 16; m <<= 1) v += __shfl_xor(v, m, 32);
  return v;
}

// ---------------- conversion kernels ----------------

// W[K][N] f32 -> Wt[N][K] f16 (K-major for WMMA B fragments)
__global__ void wt_convert_kernel(const float* __restrict__ W,
                                  _Float16* __restrict__ Wt, int K, int N) {
  size_t idx = (size_t)blockIdx.x * blockDim.x + threadIdx.x;
  size_t total = (size_t)K * N;
  if (idx >= total) return;
  int n = (int)(idx / K);
  int k = (int)(idx % K);
  Wt[idx] = (_Float16)W[(size_t)k * N + n];
}

__global__ void f32_to_f16_kernel(const float* __restrict__ in,
                                  _Float16* __restrict__ out, size_t n) {
  size_t idx = (size_t)blockIdx.x * blockDim.x + threadIdx.x;
  if (idx < n) out[idx] = (_Float16)in[idx];
}

// Vf16 [B*T, 768] head-packed -> Vt [B,H,64,T]  (T-major rows for P*V B frags)
__global__ void transpose_v_kernel(const _Float16* __restrict__ V,
                                   _Float16* __restrict__ Vt) {
  size_t idx = (size_t)blockIdx.x * blockDim.x + threadIdx.x;
  size_t total = (size_t)BB * NH * DH * TT;
  if (idx >= total) return;
  int t = (int)(idx % TT);
  size_t r = idx / TT;
  int d = (int)(r % DH); r /= DH;
  int h = (int)(r % NH);
  int b = (int)(r / NH);
  Vt[idx] = V[((size_t)(b * TT + t)) * DM + h * DH + d];
}

// ---------------- layernorm (f32 in -> f16 out) ----------------

__global__ void __launch_bounds__(256) layernorm_kernel(
    const float* __restrict__ x, const float* __restrict__ g,
    const float* __restrict__ bta, _Float16* __restrict__ out) {
  __shared__ float sred[512];
  const int row = blockIdx.x;
  const int tid = threadIdx.x;
  const float* xr = x + (size_t)row * DM;
  float v[3];
  float s = 0.f, s2 = 0.f;
#pragma unroll
  for (int i = 0; i < 3; ++i) {
    v[i] = xr[tid + i * 256];
    s += v[i];
    s2 += v[i] * v[i];
  }
  sred[tid] = s;
  sred[256 + tid] = s2;
  __syncthreads();
  for (int off = 128; off > 0; off >>= 1) {
    if (tid < off) {
      sred[tid] += sred[tid + off];
      sred[256 + tid] += sred[256 + tid + off];
    }
    __syncthreads();
  }
  const float mu = sred[0] * (1.0f / DM);
  const float var = sred[256] * (1.0f / DM) - mu * mu;
  const float rstd = rsqrtf(var + 1e-5f);
#pragma unroll
  for (int i = 0; i < 3; ++i) {
    int c = tid + i * 256;
    out[(size_t)row * DM + c] = (_Float16)((v[i] - mu) * rstd * g[c] + bta[c]);
  }
}

// ---------------- generic WMMA GEMM ----------------
// C[M,N] = act(A[M,K]f16 * Bt[N,K]f16 + bias) + residual
// block = 128 threads (4 waves); wave -> 16x64 tile; grid = (N/64, M/64)

__global__ void __launch_bounds__(128) gemm_wmma_kernel(
    const _Float16* __restrict__ A, const _Float16* __restrict__ Bt,
    const float* __restrict__ bias, const float* __restrict__ residual,
    float* __restrict__ Cf, _Float16* __restrict__ Ch,
    int M, int N, int K, int act) {
  const int lane = threadIdx.x & 31;
  const int wave = threadIdx.x >> 5;
  const int lo = lane & 15;
  const int mbase = (blockIdx.y * 4 + wave) * 16;
  const int ntile0 = blockIdx.x * 4;

  f8v acc[4];
#pragma unroll
  for (int n = 0; n < 4; ++n) acc[n] = zero8();

  const int rowA = mbase + lo;
  for (int k0 = 0; k0 < K; k0 += 32) {
    h16v a = load_a_frag(A, (size_t)K, rowA, k0, lane);
#pragma unroll
    for (int n = 0; n < 4; ++n) {
      const int col = (ntile0 + n) * 16 + lo;
      h16v b = load_bt_frag(Bt, (size_t)K, col, k0, lane);
      acc[n] = wmma_f16(a, b, acc[n]);
    }
  }

  const int rlo = (lane >> 4) * 8;  // row offset 0 or 8 within the 16-row tile
#pragma unroll
  for (int n = 0; n < 4; ++n) {
    const int col = (ntile0 + n) * 16 + lo;
    const float bv = bias ? bias[col] : 0.0f;
#pragma unroll
    for (int r = 0; r < 8; ++r) {
      const int row = mbase + rlo + r;
      float v = acc[n][r] + bv;
      if (act == 1) v = 0.5f * v * (1.0f + erff(v * 0.70710678118654752f));
      if (residual) v += residual[(size_t)row * N + col];
      if (Cf) Cf[(size_t)row * N + col] = v;
      if (Ch) Ch[(size_t)row * N + col] = (_Float16)v;
    }
  }
}

// ---------------- flash attention with Hadamard positional scores ----------------
// block = 128 (4 waves), each wave owns a 16-row q tile. grid = (T/64, B*H).
// scores = (Q K^T)(Qp Kp^T) / 64, causal, online softmax, O = P V.

__global__ void __launch_bounds__(128) attn_kernel(
    const _Float16* __restrict__ Qc, const _Float16* __restrict__ Kc,
    const _Float16* __restrict__ qp, const _Float16* __restrict__ kp,
    const _Float16* __restrict__ Vt, _Float16* __restrict__ Oout) {
  __shared__ __align__(32) _Float16 pbuf[4][16 * 32];
  const int lane = threadIdx.x & 31;
  const int wave = threadIdx.x >> 5;
  const int lo = lane & 15;
  const int hi = lane >> 4;  // 0 or 1
  const int bh = blockIdx.y;
  const int b = bh / NH, h = bh % NH;
  const int qbase = (blockIdx.x * 4 + wave) * 16;

  const _Float16* qc_base = Qc + (size_t)b * TT * DM + (size_t)h * DH;
  const _Float16* qp_base = qp + ((size_t)(b * NH + h)) * TT * DH;
  const int rowq = qbase + lo;

  h16v aq[2], ap[2];
#pragma unroll
  for (int kd = 0; kd < 2; ++kd) {
    aq[kd] = load_a_frag(qc_base, (size_t)DM, rowq, kd * 32, lane);
    ap[kd] = load_a_frag(qp_base, (size_t)DH, rowq, kd * 32, lane);
  }

  float m_i[8], l_i[8];
  f8v o[4];
#pragma unroll
  for (int r = 0; r < 8; ++r) { m_i[r] = -3.0e38f; l_i[r] = 0.0f; }
#pragma unroll
  for (int n = 0; n < 4; ++n) o[n] = zero8();

  _Float16* pb = &pbuf[wave][0];
  const int jmax = qbase + 15;

  for (int j0 = 0; j0 <= jmax; j0 += 32) {
    f8v sx[2], sp[2];
#pragma unroll
    for (int nt = 0; nt < 2; ++nt) { sx[nt] = zero8(); sp[nt] = zero8(); }
#pragma unroll
    for (int nt = 0; nt < 2; ++nt) {
      const int j = j0 + nt * 16 + lo;  // key index this lane's B column covers
#pragma unroll
      for (int kd = 0; kd < 2; ++kd) {
        // K^T fragment: contiguous 16 halves over d from row-major K
        const _Float16* kcb =
            Kc + ((size_t)(b * TT + j)) * DM + h * DH + kd * 32 + hi * 16;
        h16v bk = *(const h16v*)kcb;
        sx[nt] = wmma_f16(aq[kd], bk, sx[nt]);
        const _Float16* kpb =
            kp + (((size_t)(b * NH + h)) * TT + j) * DH + kd * 32 + hi * 16;
        h16v bp = *(const h16v*)kpb;
        sp[nt] = wmma_f16(ap[kd], bp, sp[nt]);
      }
    }

    float p[2][8];
#pragma unroll
    for (int nt = 0; nt < 2; ++nt)
#pragma unroll
      for (int r = 0; r < 8; ++r) {
        float v = sx[nt][r] * sp[nt][r] * (1.0f / 64.0f);  // /sqrt(64)^2
        const int q = qbase + hi * 8 + r;
        const int j = j0 + nt * 16 + lo;
        p[nt][r] = (j > q) ? -3.0e38f : v;
      }

    float corr[8];
#pragma unroll
    for (int r = 0; r < 8; ++r) {
      float mx = redmax16(fmaxf(p[0][r], p[1][r]));
      float mn = fmaxf(m_i[r], mx);
      corr[r] = expf(m_i[r] - mn);
      p[0][r] = expf(p[0][r] - mn);
      p[1][r] = expf(p[1][r] - mn);
      float rs = redsum16(p[0][r] + p[1][r]);
      l_i[r] = l_i[r] * corr[r] + rs;
      m_i[r] = mn;
    }
#pragma unroll
    for (int n = 0; n < 4; ++n)
#pragma unroll
      for (int r = 0; r < 8; ++r) o[n][r] *= corr[r];

    // Stage P (D-layout) -> LDS -> reload as A-fragment layout for P*V.
#pragma unroll
    for (int nt = 0; nt < 2; ++nt)
#pragma unroll
      for (int r = 0; r < 8; ++r)
        pb[(hi * 8 + r) * 32 + nt * 16 + lo] = (_Float16)p[nt][r];
    asm volatile("s_wait_dscnt 0" ::: "memory");
    h16v apv;
    {
      const _Float16* pr = pb + lo * 32 + hi * 16;
      h8v plo = *(const h8v*)pr;
      h8v phi = *(const h8v*)(pr + 8);
      apv = combine16(plo, phi);
    }
#pragma unroll
    for (int n = 0; n < 4; ++n) {
      // V^T fragment: row d = n*16+lo of Vt, contiguous over keys
      const _Float16* vb =
          Vt + (((size_t)(b * NH + h)) * DH + n * 16 + lo) * TT + j0 + hi * 16;
      h16v bv = *(const h16v*)vb;
      o[n] = wmma_f16(apv, bv, o[n]);
    }
    asm volatile("s_wait_dscnt 0" ::: "memory");
  }

#pragma unroll
  for (int n = 0; n < 4; ++n)
#pragma unroll
    for (int r = 0; r < 8; ++r) {
      const int q = qbase + hi * 8 + r;
      const int d = n * 16 + lo;
      const float v = o[n][r] / l_i[r];
      Oout[((size_t)(b * TT + q)) * DM + h * DH + d] = (_Float16)v;
    }
}

// ---------------- host orchestration ----------------

extern "C" void kernel_launch(void* const* d_in, const int* in_sizes, int n_in,
                              void* d_out, int out_size, void* d_ws, size_t ws_size,
                              hipStream_t stream) {
  (void)in_sizes; (void)n_in; (void)out_size; (void)ws_size;

  const float* x    = (const float*)d_in[0];
  const float* qpos = (const float*)d_in[1];
  const float* kpos = (const float*)d_in[2];
  // d_in[3] = causal_mask (bool) -- causality computed analytically
  const float* Wq = (const float*)d_in[4];
  const float* bq = (const float*)d_in[5];
  const float* Wk = (const float*)d_in[6];
  const float* bk = (const float*)d_in[7];
  const float* Wv = (const float*)d_in[8];
  const float* bv = (const float*)d_in[9];
  const float* Wo = (const float*)d_in[10];
  const float* bo = (const float*)d_in[11];
  const float* ln1g = (const float*)d_in[12];
  const float* ln1b = (const float*)d_in[13];
  const float* ln2g = (const float*)d_in[14];
  const float* ln2b = (const float*)d_in[15];
  const float* W1 = (const float*)d_in[16];
  const float* b1 = (const float*)d_in[17];
  const float* W2 = (const float*)d_in[18];
  const float* b2 = (const float*)d_in[19];
  float* out = (float*)d_out;

  char* ws = (char*)d_ws;
  size_t off = 0;
  auto alloc = [&](size_t bytes) -> void* {
    off = (off + 255) & ~(size_t)255;
    void* p = ws + off;
    off += bytes;
    return p;
  };

  const size_t nDD = (size_t)DM * DM;
  const size_t nDF = (size_t)DM * DFF;
  const size_t nPos = (size_t)BB * NH * TT * DH;

  _Float16* Wtq = (_Float16*)alloc(nDD * 2);
  _Float16* Wtk = (_Float16*)alloc(nDD * 2);
  _Float16* Wtv = (_Float16*)alloc(nDD * 2);
  _Float16* Wto = (_Float16*)alloc(nDD * 2);
  _Float16* Wt1 = (_Float16*)alloc(nDF * 2);  // [DFF][DM]
  _Float16* Wt2 = (_Float16*)alloc(nDF * 2);  // [DM][DFF]
  _Float16* qp16 = (_Float16*)alloc(nPos * 2);
  _Float16* kp16 = (_Float16*)alloc(nPos * 2);
  _Float16* h1  = (_Float16*)alloc((size_t)MM * DM * 2);
  _Float16* Qf  = (_Float16*)alloc((size_t)MM * DM * 2);
  _Float16* Kf  = (_Float16*)alloc((size_t)MM * DM * 2);
  _Float16* Vf  = (_Float16*)alloc((size_t)MM * DM * 2);
  _Float16* Vt  = (_Float16*)alloc((size_t)MM * DM * 2);
  _Float16* ao  = (_Float16*)alloc((size_t)MM * DM * 2);
  float*    x2  = (float*)alloc((size_t)MM * DM * 4);
  _Float16* h2  = (_Float16*)alloc((size_t)MM * DM * 2);
  _Float16* ff1 = (_Float16*)alloc((size_t)MM * DFF * 2);

  auto blks = [](size_t total) { return (unsigned)((total + 255) / 256); };

  // 1) weight conversion (transposed, f16)
  wt_convert_kernel<<<blks(nDD), 256, 0, stream>>>(Wq, Wtq, DM, DM);
  wt_convert_kernel<<<blks(nDD), 256, 0, stream>>>(Wk, Wtk, DM, DM);
  wt_convert_kernel<<<blks(nDD), 256, 0, stream>>>(Wv, Wtv, DM, DM);
  wt_convert_kernel<<<blks(nDD), 256, 0, stream>>>(Wo, Wto, DM, DM);
  wt_convert_kernel<<<blks(nDF), 256, 0, stream>>>(W1, Wt1, DM, DFF);
  wt_convert_kernel<<<blks(nDF), 256, 0, stream>>>(W2, Wt2, DFF, DM);
  f32_to_f16_kernel<<<blks(nPos), 256, 0, stream>>>(qpos, qp16, nPos);
  f32_to_f16_kernel<<<blks(nPos), 256, 0, stream>>>(kpos, kp16, nPos);

  // 2) LN1
  layernorm_kernel<<<MM, 256, 0, stream>>>(x, ln1g, ln1b, h1);

  // 3) QKV projections
  dim3 gblk(128);
  dim3 gD(DM / 64, MM / 64);
  gemm_wmma_kernel<<<gD, gblk, 0, stream>>>(h1, Wtq, bq, nullptr, nullptr, Qf,
                                            MM, DM, DM, 0);
  gemm_wmma_kernel<<<gD, gblk, 0, stream>>>(h1, Wtk, bk, nullptr, nullptr, Kf,
                                            MM, DM, DM, 0);
  gemm_wmma_kernel<<<gD, gblk, 0, stream>>>(h1, Wtv, bv, nullptr, nullptr, Vf,
                                            MM, DM, DM, 0);

  // 4) V transpose for P*V fragments
  transpose_v_kernel<<<blks((size_t)BB * NH * DH * TT), 256, 0, stream>>>(Vf, Vt);

  // 5) attention
  attn_kernel<<<dim3(TT / 64, BB * NH), 128, 0, stream>>>(Qf, Kf, qp16, kp16,
                                                          Vt, ao);

  // 6) output projection + residual -> x2 (f32)
  gemm_wmma_kernel<<<gD, gblk, 0, stream>>>(ao, Wto, bo, x, x2, nullptr,
                                            MM, DM, DM, 0);

  // 7) LN2
  layernorm_kernel<<<MM, 256, 0, stream>>>(x2, ln2g, ln2b, h2);

  // 8) FFN1 + GELU -> f16
  gemm_wmma_kernel<<<dim3(DFF / 64, MM / 64), gblk, 0, stream>>>(
      h2, Wt1, b1, nullptr, nullptr, ff1, MM, DFF, DM, 1);

  // 9) FFN2 + residual -> out (f32)
  gemm_wmma_kernel<<<gD, gblk, 0, stream>>>(ff1, Wt2, b2, x2, out, nullptr,
                                            MM, DM, DFF, 0);
}
